// CrossTemporalAttention_83064667504858
// MI455X (gfx1250) — matrix operand order repaired and checked by reading, not compiled, not run on hardware
//
#include <hip/hip_runtime.h>
#include <hip/hip_bf16.h>
#include <math.h>

typedef __attribute__((ext_vector_type(16))) _Float16 v16h;
typedef __attribute__((ext_vector_type(8)))  _Float16 v8h;
typedef __attribute__((ext_vector_type(8)))  float    v8f;

#define NTOK  4608          // 8*24*24 tokens at low res
#define TT    8
#define HH    24
#define WW    24
#define NTILE (NTOK / 16)   // 288 row tiles per (slot,batch)

// ---------------------------------------------------------------------------
// helpers
// ---------------------------------------------------------------------------
__device__ __forceinline__ v16h load16h(const _Float16* p0, const _Float16* p1) {
  v8h a = *(const v8h*)p0;
  v8h b = *(const v8h*)p1;
  v16h r;
#pragma unroll
  for (int i = 0; i < 8; ++i) { r[i] = a[i]; r[i + 8] = b[i]; }
  return r;
}

__device__ __forceinline__ float tri3(const float* __restrict__ base, int sT, int sH,
                                      int t0, int t1, int h0, int h1, int w0, int w1,
                                      float wt, float wh, float ww) {
  float v000 = base[t0 * sT + h0 * sH + w0];
  float v001 = base[t0 * sT + h0 * sH + w1];
  float v010 = base[t0 * sT + h1 * sH + w0];
  float v011 = base[t0 * sT + h1 * sH + w1];
  float v100 = base[t1 * sT + h0 * sH + w0];
  float v101 = base[t1 * sT + h0 * sH + w1];
  float v110 = base[t1 * sT + h1 * sH + w0];
  float v111 = base[t1 * sT + h1 * sH + w1];
  float a00 = v000 + (v001 - v000) * ww;
  float a01 = v010 + (v011 - v010) * ww;
  float a10 = v100 + (v101 - v100) * ww;
  float a11 = v110 + (v111 - v110) * ww;
  float b0 = a00 + (a01 - a00) * wh;
  float b1 = a10 + (a11 - a10) * wh;
  return b0 + (b1 - b0) * wt;
}

__device__ __forceinline__ void rope_apply(float* q, const float* s, const float* c) {
#pragma unroll
  for (int cp = 0; cp < 8; ++cp) {
    float x1 = q[cp], x2 = q[cp + 8];
    q[cp]     = x1 * c[cp] - x2 * s[cp];
    q[cp + 8] = x2 * c[cp] + x1 * s[cp];
  }
}

// ---------------------------------------------------------------------------
// Kernel 1: trilinear downsample (16,48,48)->(8,24,24) + 1x1 conv 64->32
// ---------------------------------------------------------------------------
__global__ void k_down(const float* __restrict__ x, const float* __restrict__ xcin,
                       const float* __restrict__ w_down, const float* __restrict__ b_down,
                       const float* __restrict__ w_down_c, const float* __restrict__ b_down_c,
                       float* __restrict__ xd, float* __restrict__ xc) {
  int gid = blockIdx.x * blockDim.x + threadIdx.x;
  if (gid >= 2 * NTOK) return;
  int b = gid / NTOK, n = gid % NTOK;
  int ww = n % WW, hh = (n / WW) % HH, tt = n / (WW * HH);

  float pT = tt * (15.f / 7.f);
  float pH = hh * (47.f / 23.f);
  float pW = ww * (47.f / 23.f);
  int t0 = (int)floorf(pT); t0 = min(max(t0, 0), 15); int t1 = min(t0 + 1, 15); float wt = pT - t0;
  int h0 = (int)floorf(pH); h0 = min(max(h0, 0), 47); int h1 = min(h0 + 1, 47); float wh = pH - h0;
  int w0 = (int)floorf(pW); w0 = min(max(w0, 0), 47); int w1 = min(w0 + 1, 47); float wz = pW - w0;

  float acc0[32], acc1[32];
#pragma unroll
  for (int co = 0; co < 32; ++co) { acc0[co] = b_down[co]; acc1[co] = b_down_c[co]; }

  for (int c = 0; c < 64; ++c) {
    const float* bx  = x    + ((size_t)b * 64 + c) * (16 * 48 * 48);
    const float* bxc = xcin + ((size_t)b * 64 + c) * (16 * 48 * 48);
    float v0 = tri3(bx,  48 * 48, 48, t0, t1, h0, h1, w0, w1, wt, wh, wz);
    float v1 = tri3(bxc, 48 * 48, 48, t0, t1, h0, h1, w0, w1, wt, wh, wz);
#pragma unroll
    for (int co = 0; co < 32; ++co) {
      acc0[co] += w_down[co * 64 + c]   * v0;
      acc1[co] += w_down_c[co * 64 + c] * v1;
    }
  }
#pragma unroll
  for (int co = 0; co < 32; ++co) {
    xd[((size_t)b * 32 + co) * NTOK + n] = acc0[co];
    xc[((size_t)b * 32 + co) * NTOK + n] = acc1[co];
  }
}

// ---------------------------------------------------------------------------
// Kernel 2: q/k/v projections + RoPE, pack to f16 WMMA-friendly layouts.
//   qh/kh: [slotb][n][32ch halves] (ch 16..31 zero pad)   q pre-scaled by 1/2
//   vh   : [slotb][32ch][NTOK halves]
//   slotb = attn_slot*2 + batch  (slot0 = cross, slot1 = self)
// ---------------------------------------------------------------------------
__global__ void k_qkv(const float* __restrict__ xd, const float* __restrict__ xc,
                      const float* __restrict__ w_q, const float* __restrict__ b_q,
                      const float* __restrict__ w_k, const float* __restrict__ b_k,
                      const float* __restrict__ w_v, const float* __restrict__ b_v,
                      const float* __restrict__ w_q_sa, const float* __restrict__ b_q_sa,
                      const float* __restrict__ w_k_sa, const float* __restrict__ b_k_sa,
                      const float* __restrict__ w_v_sa, const float* __restrict__ b_v_sa,
                      const float* __restrict__ offset, const float* __restrict__ offset_c,
                      _Float16* __restrict__ qh, _Float16* __restrict__ kh,
                      _Float16* __restrict__ vh) {
  int gid = blockIdx.x * blockDim.x + threadIdx.x;
  if (gid >= 2 * NTOK) return;
  int b = gid / NTOK, n = gid % NTOK;
  int t = n / (HH * WW);

  float xdv[32], xcv[32];
#pragma unroll
  for (int c = 0; c < 32; ++c) {
    xdv[c] = xd[((size_t)b * 32 + c) * NTOK + n];
    xcv[c] = xc[((size_t)b * 32 + c) * NTOK + n];
  }

  float sE[8], cE[8], sC[8], cC[8];
#pragma unroll
  for (int cp = 0; cp < 8; ++cp) {
    float invf = powf(10000.f, -(2.f * cp) / 16.f);
    float ang = (float)t * invf;
    float s = sinf(ang), c = cosf(ang);
    float oE = offset[cp * 500 + t];
    float oC = offset_c[cp * 500 + t];
    sE[cp] = s + oE; cE[cp] = c + oE;
    sC[cp] = s + oC; cC[cp] = c + oC;
  }

  auto mv16 = [](const float* w, const float* bia, const float* in, float* out) {
#pragma unroll
    for (int o = 0; o < 16; ++o) {
      float a = bia[o];
      for (int c = 0; c < 32; ++c) a += w[o * 32 + c] * in[c];
      out[o] = a;
    }
  };

  auto putQK = [&](_Float16* dst, int slotb, const float* v, float scale) {
    _Float16* row = dst + ((size_t)slotb * NTOK + n) * 32;
#pragma unroll
    for (int i = 0; i < 16; ++i) row[i] = (_Float16)(v[i] * scale);
#pragma unroll
    for (int i = 16; i < 32; ++i) row[i] = (_Float16)0.f;
  };

  float tmp[16];
  // cross attention (slot 0): q from x_down (ropeE), k from xc (ropeC), v from xc
  mv16(w_q, b_q, xdv, tmp); rope_apply(tmp, sE, cE); putQK(qh, b, tmp, 0.5f);
  mv16(w_k, b_k, xcv, tmp); rope_apply(tmp, sC, cC); putQK(kh, b, tmp, 1.0f);
#pragma unroll
  for (int o = 0; o < 32; ++o) {
    float a = b_v[o];
    for (int c = 0; c < 32; ++c) a += w_v[o * 32 + c] * xcv[c];
    vh[((size_t)b * 32 + o) * NTOK + n] = (_Float16)a;
  }
  // self attention (slot 1): all from xc, ropeE
  mv16(w_q_sa, b_q_sa, xcv, tmp); rope_apply(tmp, sE, cE); putQK(qh, 2 + b, tmp, 0.5f);
  mv16(w_k_sa, b_k_sa, xcv, tmp); rope_apply(tmp, sE, cE); putQK(kh, 2 + b, tmp, 1.0f);
#pragma unroll
  for (int o = 0; o < 32; ++o) {
    float a = b_v_sa[o];
    for (int c = 0; c < 32; ++c) a += w_v_sa[o * 32 + c] * xcv[c];
    vh[((size_t)(2 + b) * 32 + o) * NTOK + n] = (_Float16)a;
  }
}

// ---------------------------------------------------------------------------
// Kernel 3: flash attention, one wave per 16-query tile, 8 waves per block all
// on the same (slot,batch) so K/V chunks are staged ONCE per block into LDS
// with async global->LDS b128 copies (ASYNCcnt), double buffered against the
// WMMA compute on the previous chunk. Operands then come from ds_load_b128.
//   S^T tiles via v_wmma_f32_16x16x32_f16 (K = 32 zero-padded channels),
//   online softmax with per-lane column stats (shfl_xor(16) cross-half),
//   O^T accumulated with two more WMMAs per 32-key chunk.
//   att out: [slotb][32ch][NTOK] f32.
// ---------------------------------------------------------------------------
__global__ void __launch_bounds__(256) k_attn(const _Float16* __restrict__ qh,
                                              const _Float16* __restrict__ kh,
                                              const _Float16* __restrict__ vh,
                                              float* __restrict__ att) {
  __shared__ _Float16 kbuf[2][32][32];   // [buf][key row][32 ch halves]   2KB/buf
  __shared__ _Float16 vbuf[2][32][32];   // [buf][ch row][32 key halves]   2KB/buf

  const int tid  = threadIdx.x;
  const int lane = tid & 31;
  const int wid  = blockIdx.x * (blockDim.x >> 5) + (tid >> 5);   // grid is exact
  const int slotb = wid / NTILE;
  const int n0 = (wid % NTILE) * 16;
  const int hi = lane >> 4;
  const int nl = lane & 15;
  const int base = hi * 8;     // A-operand K sub-block start
  const v8f zc = {0.f, 0.f, 0.f, 0.f, 0.f, 0.f, 0.f, 0.f};

  const _Float16* kslot = kh + (size_t)slotb * NTOK * 32;
  const _Float16* vslot = vh + (size_t)slotb * 32 * NTOK;

  // -------- async staging setup: one b128 (16B) per thread per chunk -------
  // threads 0..127 stage K (32 rows x 64B), threads 128..255 stage V.
  const bool isK = tid < 128;
  const int  srow = (tid & 127) >> 2;   // 0..31
  const int  spart = tid & 3;           // 16B quarter of a 64B row
  const unsigned lds_kbase = (unsigned)(size_t)&kbuf[0][0][0];
  const unsigned lds_vbase = (unsigned)(size_t)&vbuf[0][0][0];
  const unsigned lds_local = (unsigned)(srow * 64 + spart * 16);
  const unsigned long long gbase =
      isK ? (unsigned long long)(size_t)kslot : (unsigned long long)(size_t)vslot;
  const unsigned stage_base = (isK ? lds_kbase : lds_vbase) + lds_local;

  auto stage = [&](int buf, int m0) {
    unsigned ldsoff = stage_base + (unsigned)buf * 2048u;
    unsigned goff = isK ? (unsigned)((m0 + srow) * 64 + spart * 16)
                        : (unsigned)(srow * (NTOK * 2) + m0 * 2 + spart * 16);
    asm volatile("global_load_async_to_lds_b128 %0, %1, %2"
                 :: "v"(ldsoff), "v"(goff), "s"(gbase)
                 : "memory");
  };

  // Q as B-operand: lane holds column n0+nl, rows (channels) hi*8..+7, 16+hi*8..+7
  const _Float16* qrow = qh + ((size_t)slotb * NTOK + n0 + nl) * 32 + hi * 8;
  v16h bq = load16h(qrow, qrow + 16);

  v8f o0 = zc, o1 = zc;
  float m_run = -1e30f, l_run = 0.f;

  stage(0, 0);   // prefetch first chunk

  for (int chunk = 0; chunk < NTOK / 32; ++chunk) {
    const int buf = chunk & 1;
    const int next = (chunk + 1) % (NTOK / 32);  // wrap: harmless re-stage at end
    stage(buf ^ 1, next * 32);

    // own previous stage complete (<=1 leaves the just-issued one in flight,
    // which targets the other buffer), then block-wide handshake.
    asm volatile("s_wait_asynccnt 0x1" ::: "memory");
    __syncthreads();

    // K tiles as A-operand from LDS: rows = keys,
    // K-dim = channels {0..7,16..23} | {8..15,24..31}
    const _Float16* kr0 = &kbuf[buf][nl][0];
    const _Float16* kr1 = &kbuf[buf][16 + nl][0];
    v16h ak0 = load16h(kr0 + base, kr0 + base + 16);
    v16h ak1 = load16h(kr1 + base, kr1 + base + 16);
    v8f s0 = __builtin_amdgcn_wmma_f32_16x16x32_f16(false, ak0, false, bq, (short)0, zc, false, false);
    v8f s1 = __builtin_amdgcn_wmma_f32_16x16x32_f16(false, ak1, false, bq, (short)0, zc, false, false);

    // column (query) max over 32 keys: 16 regs in-lane + cross-half shuffle
    float cmax = s0[0];
#pragma unroll
    for (int i = 1; i < 8; ++i) cmax = fmaxf(cmax, s0[i]);
#pragma unroll
    for (int i = 0; i < 8; ++i) cmax = fmaxf(cmax, s1[i]);
    cmax = fmaxf(cmax, __shfl_xor(cmax, 16, 32));
    float mnew = fmaxf(m_run, cmax);
    float rescale = __expf(m_run - mnew);

    float p0[8], p1[8], psum = 0.f;
#pragma unroll
    for (int i = 0; i < 8; ++i) {
      p0[i] = __expf(s0[i] - mnew);
      p1[i] = __expf(s1[i] - mnew);
      psum += p0[i] + p1[i];
    }
    psum += __shfl_xor(psum, 16, 32);
    l_run = l_run * rescale + psum;
    m_run = mnew;
#pragma unroll
    for (int i = 0; i < 8; ++i) { o0[i] *= rescale; o1[i] *= rescale; }

    // P^T as B-operand (32 keys x 16 queries): built in-lane from p0/p1, no exchange
    v16h bp;
#pragma unroll
    for (int i = 0; i < 8; ++i) {
      float x0 = __shfl_xor(p0[i], 16, 32);
      float x1 = __shfl_xor(p1[i], 16, 32);
      bp[i]     = (_Float16)p0[i];   // keys 8*hi+0..7
      bp[i + 8] = (_Float16)p1[i];   // keys 16+8*hi+0..7
    }

    // V tiles as A-operand from LDS: rows = channels, K-dim = 32 keys
    const _Float16* vr0 = &vbuf[buf][nl][0];
    const _Float16* vr1 = &vbuf[buf][16 + nl][0];
    v16h av0 = load16h(vr0 + base, vr0 + base + 16);
    v16h av1 = load16h(vr1 + base, vr1 + base + 16);
    o0 = __builtin_amdgcn_wmma_f32_16x16x32_f16(false, av0, false, bp, (short)0, o0, false, false);
    o1 = __builtin_amdgcn_wmma_f32_16x16x32_f16(false, av1, false, bp, (short)0, o1, false, false);

    // all waves done reading this buffer before it is restaged next iteration
    __syncthreads();
  }

  float inv = 1.0f / l_run;
  float* ab = att + (size_t)slotb * 32 * NTOK + n0 + nl;
#pragma unroll
  for (int i = 0; i < 8; ++i) {
    int c0 = i + 8 * hi;
    ab[(size_t)c0 * NTOK]        = o0[i] * inv;
    ab[(size_t)(16 + c0) * NTOK] = o1[i] * inv;
  }
}

// ---------------------------------------------------------------------------
// Kernel 4: groupnorm statistics per (slot,batch) over 32*NTOK values
// ---------------------------------------------------------------------------
__global__ void k_gnstats(const float* __restrict__ att, float* __restrict__ stats) {
  int sb = blockIdx.x;
  const float* p = att + (size_t)sb * 32 * NTOK;
  float s1 = 0.f, s2 = 0.f;
  for (int i = threadIdx.x; i < 32 * NTOK; i += blockDim.x) {
    float v = p[i];
    s1 += v; s2 += v * v;
  }
  __shared__ float sh1[256], sh2[256];
  sh1[threadIdx.x] = s1; sh2[threadIdx.x] = s2;
  __syncthreads();
  for (int s = 128; s > 0; s >>= 1) {
    if ((int)threadIdx.x < s) {
      sh1[threadIdx.x] += sh1[threadIdx.x + s];
      sh2[threadIdx.x] += sh2[threadIdx.x + s];
    }
    __syncthreads();
  }
  if (threadIdx.x == 0) {
    float inv = 1.f / (32.f * NTOK);
    float mu = sh1[0] * inv;
    float var = sh2[0] * inv - mu * mu;
    stats[sb * 2 + 0] = mu;
    stats[sb * 2 + 1] = rsqrtf(var + 1e-5f);
  }
}

// ---------------------------------------------------------------------------
// Kernel 5: groupnorm affine + 1x1 conv 32->64 at low res
// ---------------------------------------------------------------------------
__global__ void k_up(const float* __restrict__ att, const float* __restrict__ stats,
                     const float* __restrict__ gn_w, const float* __restrict__ gn_b,
                     const float* __restrict__ gn_w_sa, const float* __restrict__ gn_b_sa,
                     const float* __restrict__ w_up, const float* __restrict__ b_up,
                     const float* __restrict__ w_up_sa, const float* __restrict__ b_up_sa,
                     float* __restrict__ ylow) {
  int gid = blockIdx.x * blockDim.x + threadIdx.x;
  if (gid >= 4 * NTOK) return;
  int sb = gid / NTOK, n = gid % NTOK;
  int slot = sb >> 1;
  const float* gw = slot ? gn_w_sa : gn_w;
  const float* gb = slot ? gn_b_sa : gn_b;
  const float* wu = slot ? w_up_sa : w_up;
  const float* bu = slot ? b_up_sa : b_up;
  float mu = stats[sb * 2], rs = stats[sb * 2 + 1];
  float xn[32];
#pragma unroll
  for (int c = 0; c < 32; ++c)
    xn[c] = (att[((size_t)sb * 32 + c) * NTOK + n] - mu) * rs * gw[c] + gb[c];
#pragma unroll 4
  for (int co = 0; co < 64; ++co) {
    float a = bu[co];
    for (int c = 0; c < 32; ++c) a += wu[co * 32 + c] * xn[c];
    ylow[((size_t)sb * 64 + co) * NTOK + n] = a;
  }
}

// ---------------------------------------------------------------------------
// Kernel 6: trilinear upsample (8,24,24)->(16,48,48), combine
//   out = up(self) + x_c - 0.5 * up(cross)
// ---------------------------------------------------------------------------
__global__ void k_final(const float* __restrict__ ylow, const float* __restrict__ x_c,
                        float* __restrict__ out) {
  int gid = blockIdx.x * blockDim.x + threadIdx.x;
  if (gid >= 2 * 64 * 16 * 48 * 48) return;
  int w = gid % 48;
  int h = (gid / 48) % 48;
  int t = (gid / 2304) % 16;
  int c = (gid / 36864) % 64;
  int b = gid / 2359296;

  float pT = t * (7.f / 15.f);
  float pH = h * (23.f / 47.f);
  float pW = w * (23.f / 47.f);
  int t0 = (int)floorf(pT); t0 = min(max(t0, 0), 7);  int t1 = min(t0 + 1, 7);  float wt = pT - t0;
  int h0 = (int)floorf(pH); h0 = min(max(h0, 0), 23); int h1 = min(h0 + 1, 23); float wh = pH - h0;
  int w0 = (int)floorf(pW); w0 = min(max(w0, 0), 23); int w1 = min(w0 + 1, 23); float wz = pW - w0;

  const float* base0 = ylow + ((size_t)(b) * 64 + c) * NTOK;        // cross slot
  const float* base1 = ylow + ((size_t)(2 + b) * 64 + c) * NTOK;    // self slot
  float up0 = tri3(base0, HH * WW, WW, t0, t1, h0, h1, w0, w1, wt, wh, wz);
  float up1 = tri3(base1, HH * WW, WW, t0, t1, h0, h1, w0, w1, wt, wh, wz);
  out[gid] = up1 + x_c[gid] - 0.5f * up0;
}

// ---------------------------------------------------------------------------
extern "C" void kernel_launch(void* const* d_in, const int* in_sizes, int n_in,
                              void* d_out, int out_size, void* d_ws, size_t ws_size,
                              hipStream_t stream) {
  const float* x        = (const float*)d_in[0];
  const float* x_c      = (const float*)d_in[1];
  const float* w_down   = (const float*)d_in[2];
  const float* b_down   = (const float*)d_in[3];
  const float* w_down_c = (const float*)d_in[4];
  const float* b_down_c = (const float*)d_in[5];
  const float* w_q      = (const float*)d_in[6];
  const float* b_q      = (const float*)d_in[7];
  const float* w_k      = (const float*)d_in[8];
  const float* b_k      = (const float*)d_in[9];
  const float* w_v      = (const float*)d_in[10];
  const float* b_v      = (const float*)d_in[11];
  const float* w_q_sa   = (const float*)d_in[12];
  const float* b_q_sa   = (const float*)d_in[13];
  const float* w_k_sa   = (const float*)d_in[14];
  const float* b_k_sa   = (const float*)d_in[15];
  const float* w_v_sa   = (const float*)d_in[16];
  const float* b_v_sa   = (const float*)d_in[17];
  const float* gn_w     = (const float*)d_in[18];
  const float* gn_b     = (const float*)d_in[19];
  const float* gn_w_sa  = (const float*)d_in[20];
  const float* gn_b_sa  = (const float*)d_in[21];
  const float* w_up     = (const float*)d_in[22];
  const float* b_up     = (const float*)d_in[23];
  const float* w_up_sa  = (const float*)d_in[24];
  const float* b_up_sa  = (const float*)d_in[25];
  const float* offset   = (const float*)d_in[26];
  const float* offset_c = (const float*)d_in[27];

  char* ws = (char*)d_ws;
  size_t off = 0;
  auto carve = [&](size_t bytes) -> void* {
    void* p = ws + off;
    off = (off + bytes + 255) & ~(size_t)255;
    return p;
  };
  float*    xd    = (float*)carve((size_t)2 * 32 * NTOK * 4);
  float*    xc    = (float*)carve((size_t)2 * 32 * NTOK * 4);
  _Float16* qh    = (_Float16*)carve((size_t)4 * NTOK * 32 * 2);
  _Float16* kh    = (_Float16*)carve((size_t)4 * NTOK * 32 * 2);
  _Float16* vh    = (_Float16*)carve((size_t)4 * 32 * NTOK * 2);
  float*    att   = (float*)carve((size_t)4 * 32 * NTOK * 4);
  float*    stats = (float*)carve(8 * 4);
  float*    ylow  = (float*)carve((size_t)4 * 64 * NTOK * 4);

  k_down<<<36, 256, 0, stream>>>(x, x_c, w_down, b_down, w_down_c, b_down_c, xd, xc);
  k_qkv<<<36, 256, 0, stream>>>(xd, xc, w_q, b_q, w_k, b_k, w_v, b_v,
                                w_q_sa, b_q_sa, w_k_sa, b_k_sa, w_v_sa, b_v_sa,
                                offset, offset_c, qh, kh, vh);
  k_attn<<<144, 256, 0, stream>>>(qh, kh, vh, att);            // 1152 waves = 4*288 tiles
  k_gnstats<<<4, 256, 0, stream>>>(att, stats);
  k_up<<<72, 256, 0, stream>>>(att, stats, gn_w, gn_b, gn_w_sa, gn_b_sa,
                               w_up, b_up, w_up_sa, b_up_sa, ylow);
  k_final<<<18432, 256, 0, stream>>>(ylow, x_c, (float*)d_out);
}
